// Decoder_32074815767263
// MI455X (gfx1250) — matrix operand order repaired and checked by reading, not compiled, not run on hardware
//
#include <hip/hip_runtime.h>

// Problem constants (from reference)
#define B_  4
#define T_  300
#define U_  60
#define D_  512      // ENC_D == DEC_D
#define H_  1024     // INNER_DIM
#define V_  1000     // VOCAB
#define NT_ 63       // ceil(V/16) N-tiles
#define KS_ 32       // H/32 K-steps (bf16 WMMA K=32)

typedef __bf16 bf16x16 __attribute__((ext_vector_type(16)));
typedef float  v8f     __attribute__((ext_vector_type(8)));
typedef float  f32x4   __attribute__((ext_vector_type(4)));
typedef float  v2f     __attribute__((ext_vector_type(2)));
typedef int    v4i_    __attribute__((ext_vector_type(4)));

#if defined(__has_builtin)
#if __has_builtin(__builtin_amdgcn_global_load_async_to_lds_b128) && \
    __has_builtin(__builtin_amdgcn_s_wait_asynccnt)
#define HAVE_ASYNC_LDS 1
#endif
#endif

#ifdef HAVE_ASYNC_LDS
// async copy of 16 aligned bytes: global -> LDS, tracked by ASYNCcnt.
// Builtin signature (per hipcc diagnostic): (int4 AS1*, int4 AS3*, Ii, Ii)
__device__ __forceinline__ void async_cp16(const float* g, float* l) {
  __builtin_amdgcn_global_load_async_to_lds_b128(
      (__attribute__((address_space(1))) v4i_*)g,
      (__attribute__((address_space(3))) v4i_*)l,
      0, 0);
}
#endif

__device__ __forceinline__ __bf16 f2bf(float f) {
  // round-to-nearest-even f32 -> bf16
  unsigned u = __float_as_uint(f);
  u += 0x7FFFu + ((u >> 16) & 1u);
  unsigned short s = (unsigned short)(u >> 16);
  __bf16 r;
  __builtin_memcpy(&r, &s, sizeof(r));
  return r;
}

__device__ __forceinline__ float fast_tanh(float x) {
  // tanh(x) = 1 - 2/(exp(2x)+1); saturates correctly at +/-inf
  float e = __expf(2.f * x);
  return 1.f - 2.f / (e + 1.f);
}

// ---------------------------------------------------------------------------
// Projections on the f32 matrix pipe (V_WMMA_F32_16X16X4_F32):
//   encp[row][n] = enc_state[row][:] @ W1[0:512][n]   + b1[n]   (blocks 0..74)
//   decp[row][n] = dec_state[row][:] @ W1[512:1024][n]          (blocks 75..89)
// One 16-row M-tile per block; 8 waves x 8 N-tiles cover N = 1024.
// K loop: 128 steps of 4.
// A-frag (f32 16x4): lanes 0-15 M=0-15 K={0,1}; lanes 16-31 K={2,3}.
// B-frag (f32 4x16): lanes 0-15 N,K={0,1};     lanes 16-31 K={2,3}.
// ---------------------------------------------------------------------------
__global__ __launch_bounds__(256) void proj_wmma_kernel(
    const float* __restrict__ enc,  // [1200][512]
    const float* __restrict__ dec,  // [240][512]
    const float* __restrict__ W1,   // [1024][1024] row-major
    const float* __restrict__ b1,   // [1024]
    float* __restrict__ encp,       // [1200][1024]
    float* __restrict__ decp)       // [240][1024]
{
  __shared__ __attribute__((aligned(16)))
      float s_x[16][D_ + 4];   // +4 pad: stride 516 floats -> conflict-free,
                               // still 16B-aligned rows (2064 B)
  const int tid = threadIdx.x;

  const bool is_enc = blockIdx.x < (B_ * T_) / 16;
  const int  mtile  = is_enc ? blockIdx.x : blockIdx.x - (B_ * T_) / 16;
  const int  row0   = mtile * 16;
  const float* X = is_enc ? enc : dec;
  float*       Y = is_enc ? encp : decp;
  const int w1_row_off = is_enc ? 0 : D_;

#ifdef HAVE_ASYNC_LDS
  // 16 rows x 512 f32 = 2048 16-byte chunks, 8 per thread, ASYNCcnt-tracked
  #pragma unroll
  for (int it = 0; it < 8; ++it) {
    int c = it * 256 + tid;              // chunk id
    int m = c >> 7, off = (c & 127) * 4;
    async_cp16(X + (size_t)(row0 + m) * D_ + off, &s_x[m][off]);
  }
  __builtin_amdgcn_s_wait_asynccnt(0);
#else
  for (int i = tid; i < 16 * D_; i += 256) {
    int m = i >> 9, d = i & (D_ - 1);
    s_x[m][d] = X[(size_t)(row0 + m) * D_ + d];
  }
#endif
  __syncthreads();

  const int wave  = tid >> 5;
  const int lane  = tid & 31;
  const int m     = lane & 15;
  const int khalf = (lane >> 4) * 2;  // 0 or 2

  v8f acc[8] = {};
  const float* wbase = W1 + (size_t)w1_row_off * H_ + (lane & 15) + wave * 128;

  for (int ks = 0; ks < D_ / 4; ++ks) {
    int k = ks * 4 + khalf;
    v2f a;
    a.x = s_x[m][k];
    a.y = s_x[m][k + 1];
    const float* wb = wbase + (size_t)k * H_;
    #pragma unroll
    for (int j = 0; j < 8; ++j) {
      v2f bfrag;
      bfrag.x = wb[j * 16];
      bfrag.y = wb[j * 16 + H_];
      acc[j] = __builtin_amdgcn_wmma_f32_16x16x4_f32(
          false, a, false, bfrag, (short)0, acc[j], false, false);
    }
  }

  // D layout: vgpr r, lanes 0-15 -> M=r, lanes 16-31 -> M=8+r; N = lane&15.
  const int mrow = (lane >> 4) * 8;
  #pragma unroll
  for (int j = 0; j < 8; ++j) {
    int n = wave * 128 + j * 16 + (lane & 15);
    float bias = is_enc ? b1[n] : 0.f;
    #pragma unroll
    for (int r = 0; r < 8; ++r)
      Y[(size_t)(row0 + mrow + r) * H_ + n] = acc[j][r] + bias;
  }
}

// ---------------------------------------------------------------------------
// Repack W2 (f32 [1024][1000]) into bf16 WMMA B-fragment layout:
//   w2f[nt][ks][lane][h],  K = ks*32 + (lane&16 ? 16:0) + h,  N = nt*16 + (lane&15)
// Each lane's 16 halves for one fragment are contiguous (32 bytes).
// Total 63*32*512 bf16 = 2.06 MB -> fully L2-resident.
// ---------------------------------------------------------------------------
__global__ __launch_bounds__(256) void w2frag_kernel(
    const float* __restrict__ W2, __bf16* __restrict__ w2f)
{
  int flat = blockIdx.x * 256 + threadIdx.x;
  if (flat >= NT_ * KS_ * 32 * 16) return;
  int h    = flat & 15;
  int lane = (flat >> 4) & 31;
  int ks   = (flat >> 9) & 31;
  int nt   = flat >> 14;
  int k = ks * 32 + ((lane & 16) ? 16 : 0) + h;
  int n = nt * 16 + (lane & 15);
  float v = (n < V_) ? W2[(size_t)k * V_ + n] : 0.f;
  w2f[flat] = f2bf(v);
}

// ---------------------------------------------------------------------------
// Main kernel: one workgroup per (b,t).  Builds hidden = tanh(enc+dec) for
// 64 padded u-rows as bf16 in LDS (WMMA A-fragment layout), then each wave
// computes a stripe of N-tiles: C[4 u-subtiles][16 cols] accumulated over 32
// K-steps with v_wmma_f32_16x16x32_bf16 (4x A-reuse per B fragment load).
// ---------------------------------------------------------------------------
__global__ __launch_bounds__(256, 1) void joint_kernel(
    const float* __restrict__ encp,  // [B*T][1024]   (b1 folded in)
    const float* __restrict__ decp,  // [B*U][1024]
    const __bf16* __restrict__ w2f,  // [63][32][32][16] bf16
    const float* __restrict__ b2,    // [1000]
    float* __restrict__ out)         // [B*T*U][1000] f32
{
  // A-fragment staging: Ah[mt][ks][lane][h] halves  (4*32*32*16*2 = 128 KB)
  __shared__ __attribute__((aligned(128))) __bf16 Ah[4 * KS_ * 32 * 16];
  __shared__ __attribute__((aligned(16))) float s_enc[H_];

  const int wg  = blockIdx.x;         // b*T + t
  const int bb  = wg / T_;
  const int tid = threadIdx.x;

#ifdef HAVE_ASYNC_LDS
  // 4 KB row: 256 threads x 16 B, async to LDS (no VGPR round trip)
  async_cp16(encp + (size_t)wg * H_ + tid * 4, &s_enc[tid * 4]);
  __builtin_amdgcn_s_wait_asynccnt(0);
#else
  for (int i = tid; i < H_; i += 256) s_enc[i] = encp[(size_t)wg * H_ + i];
#endif
  __syncthreads();

  // hidden -> LDS in A-fragment layout.
  // A order used here (contiguous k, same as w2f): lane = m + (k&16 ? 16:0); h = k&15
  const float* decb = decp + (size_t)bb * U_ * H_;
  #pragma unroll 4
  for (int it = 0; it < (64 * H_) / 256; ++it) {
    int flat = it * 256 + tid;
    int u = flat >> 10;
    int k = flat & (H_ - 1);
    float hv = 0.f;
    if (u < U_) hv = fast_tanh(s_enc[k] + decb[(size_t)u * H_ + k]);
    int mt = u >> 4, m = u & 15;
    int ks = k >> 5, kk = k & 31;
    int lane = m + ((kk & 16) ? 16 : 0);
    int h    = kk & 15;
    Ah[(((mt * KS_) + ks) * 32 + lane) * 16 + h] = f2bf(hv);
  }
  __syncthreads();

  const int wave = tid >> 5;
  const int lane = tid & 31;
  const __bf16* abase = Ah + lane * 16;           // + (mt*KS_+ks)*512 per frag

  for (int nt = wave; nt < NT_; nt += 8) {
    v8f c0 = {}, c1 = {}, c2 = {}, c3 = {};
    const __bf16* bptr = w2f + (size_t)nt * (KS_ * 512) + lane * 16;
    #pragma unroll 2
    for (int ks = 0; ks < KS_; ++ks) {
      bf16x16 bf = *(const bf16x16*)(bptr + ks * 512);
      bf16x16 a0 = *(const bf16x16*)(abase + (0 * KS_ + ks) * 512);
      bf16x16 a1 = *(const bf16x16*)(abase + (1 * KS_ + ks) * 512);
      bf16x16 a2 = *(const bf16x16*)(abase + (2 * KS_ + ks) * 512);
      bf16x16 a3 = *(const bf16x16*)(abase + (3 * KS_ + ks) * 512);
      c0 = __builtin_amdgcn_wmma_f32_16x16x32_bf16(false, a0, false, bf, (short)0, c0, false, false);
      c1 = __builtin_amdgcn_wmma_f32_16x16x32_bf16(false, a1, false, bf, (short)0, c1, false, false);
      c2 = __builtin_amdgcn_wmma_f32_16x16x32_bf16(false, a2, false, bf, (short)0, c2, false, false);
      c3 = __builtin_amdgcn_wmma_f32_16x16x32_bf16(false, a3, false, bf, (short)0, c3, false, false);
    }

    // Epilogue.  C/D layout: vgpr r, lanes 0-15 -> M=r, lanes 16-31 -> M=8+r;
    // N = lane&15.
    int n = nt * 16 + (lane & 15);
    if (n >= V_) continue;                    // divergence only around stores
    float bias = b2[n];
    int mrow = (lane >> 4) * 8;               // 0 or 8
    size_t outb = ((size_t)wg * U_) * V_ + n;
    #pragma unroll
    for (int r = 0; r < 8; ++r) {
      int m = mrow + r;
      out[outb + (size_t)(m)      * V_] = c0[r] + bias;   // u =  0..15
      out[outb + (size_t)(16 + m) * V_] = c1[r] + bias;   // u = 16..31
      out[outb + (size_t)(32 + m) * V_] = c2[r] + bias;   // u = 32..47
      if (48 + m < U_)
        out[outb + (size_t)(48 + m) * V_] = c3[r] + bias; // u = 48..59
    }
  }
}

// ---------------------------------------------------------------------------
extern "C" void kernel_launch(void* const* d_in, const int* in_sizes, int n_in,
                              void* d_out, int out_size, void* d_ws, size_t ws_size,
                              hipStream_t stream) {
  (void)in_sizes; (void)n_in; (void)out_size; (void)ws_size;
  const float* enc = (const float*)d_in[0];   // [4][300][512]
  const float* dec = (const float*)d_in[1];   // [4][60][512]
  const float* W1  = (const float*)d_in[2];   // [1024][1024]
  const float* b1  = (const float*)d_in[3];   // [1024]
  const float* W2  = (const float*)d_in[4];   // [1024][1000]
  const float* b2  = (const float*)d_in[5];   // [1000]
  float* out = (float*)d_out;                 // [4][300][60][1000]

  // Workspace: encp (4.9 MB) | decp (0.98 MB) | w2f bf16 frags (2.06 MB)
  float*  encp = (float*)d_ws;
  float*  decp = encp + (size_t)B_ * T_ * H_;
  __bf16* w2f  = (__bf16*)(decp + (size_t)B_ * U_ * H_);

  proj_wmma_kernel<<<(B_ * T_) / 16 + (B_ * U_) / 16, 256, 0, stream>>>(
      enc, dec, W1, b1, encp, decp);
  w2frag_kernel<<<(NT_ * KS_ * 32 * 16 + 255) / 256, 256, 0, stream>>>(W2, w2f);
  joint_kernel<<<B_ * T_, 256, 0, stream>>>(encp, decp, w2f, b2, out);
}
